// HKRPQParallelBlock_17523466567933
// MI455X (gfx1250) — matrix operand order repaired and checked by reading, not compiled, not run on hardware
//
#include <hip/hip_runtime.h>
#include <stdint.h>

// ---------- types ----------
typedef __attribute__((ext_vector_type(16))) __bf16 bf16x16;
typedef __attribute__((ext_vector_type(8)))  float  f32x8;

__device__ __forceinline__ uint16_t f2bf(float x) {
    uint32_t u = __float_as_uint(x);
    uint32_t r = (u + 0x7FFFu + ((u >> 16) & 1u)) >> 16;
    return (uint16_t)r;
}

__device__ __forceinline__ bf16x16 ld_frag(const uint16_t* p0, const uint16_t* p1) {
    union { struct { uint4 lo; uint4 hi; } u; bf16x16 v; } f;
    f.u.lo = *(const uint4*)p0;
    f.u.hi = *(const uint4*)p1;
    return f.v;
}

__device__ __forceinline__ f32x8 zerof8() {
    f32x8 z = {0.f, 0.f, 0.f, 0.f, 0.f, 0.f, 0.f, 0.f};
    return z;
}

__device__ __forceinline__ f32x8 wmma_bf16(bf16x16 a, bf16x16 b, f32x8 c) {
    return __builtin_amdgcn_wmma_f32_16x16x32_bf16(false, a, false, b, (short)0, c, false, false);
}

// ---------- problem constants ----------
#define DIMC    1024
#define HEADS   16
#define HD      64
#define MLPC    4096
#define IN_OUT  7168   // MLP + 3*DIM
#define OUT_IN  5120   // MLP + DIM
#define OUT_OUT 2048
#define NTOK    4096   // B*N
#define SEQ     1024

// ================= decode Win: (7168,1024) bf16 =================
__global__ __launch_bounds__(256) void decode_win_kernel(
    const float* __restrict__ cb, const int* __restrict__ idx, uint16_t* __restrict__ W) {
    int o = blockIdx.x;            // output row 0..7167
    int j = threadIdx.x * 4;       // col 0..1023
    int c = j >> 6;                // codebook (SUB_IN=64)
    int s = j & 63;
    int cluster = o >> 6;          // 7168/112 = 64
    int i0 = idx[(size_t)(0 * IN_OUT + o) * 16 + c];
    int i1 = idx[(size_t)(1 * IN_OUT + o) * 16 + c];
    const float* p0 = cb + ((((size_t)(0 * 112 + cluster) * 16 + c) * 64 + i0) * 64 + s);
    const float* p1 = cb + ((((size_t)(1 * 112 + cluster) * 16 + c) * 64 + i1) * 64 + s);
    float4 a = *(const float4*)p0;
    float4 b = *(const float4*)p1;
    union { uint16_t h[4]; uint2 u; } r;
    r.h[0] = f2bf(a.x + b.x); r.h[1] = f2bf(a.y + b.y);
    r.h[2] = f2bf(a.z + b.z); r.h[3] = f2bf(a.w + b.w);
    *(uint2*)&W[(size_t)o * DIMC + j] = r.u;
}

// ================= decode Wout: (2048,5120) bf16 =================
__global__ __launch_bounds__(256) void decode_wout_kernel(
    const float* __restrict__ cb, const int* __restrict__ idx, uint16_t* __restrict__ W) {
    int o = blockIdx.x;             // 0..2047
    int t = threadIdx.x;
    int c = t >> 4;                 // 16 threads per codebook
    int s = (t & 15) * 20;          // SUB_OUT=320, 16*20=320
    int cluster = o >> 7;           // 2048/16 = 128
    int i0 = idx[(size_t)(0 * OUT_OUT + o) * 16 + c];
    int i1 = idx[(size_t)(1 * OUT_OUT + o) * 16 + c];
    const float* p0 = cb + ((((size_t)(0 * 16 + cluster) * 16 + c) * 64 + i0) * 320 + s);
    const float* p1 = cb + ((((size_t)(1 * 16 + cluster) * 16 + c) * 64 + i1) * 320 + s);
    uint16_t* dst = W + (size_t)o * OUT_IN + c * 320 + s;
    #pragma unroll
    for (int g = 0; g < 5; ++g) {
        float4 a = *(const float4*)(p0 + g * 4);
        float4 b = *(const float4*)(p1 + g * 4);
        union { uint16_t h[4]; uint2 u; } r;
        r.h[0] = f2bf(a.x + b.x); r.h[1] = f2bf(a.y + b.y);
        r.h[2] = f2bf(a.z + b.z); r.h[3] = f2bf(a.w + b.w);
        *(uint2*)(dst + g * 4) = r.u;
    }
}

// ================= LayerNorm -> bf16 =================
__global__ __launch_bounds__(256) void ln_kernel(const float* __restrict__ x,
                                                 uint16_t* __restrict__ xn) {
    __shared__ float red0[8], red1[8];
    int row = blockIdx.x;
    const float* xr = x + (size_t)row * DIMC;
    int t = threadIdx.x;
    float4 xv = *(const float4*)&xr[t * 4];
    float s  = xv.x + xv.y + xv.z + xv.w;
    float s2 = xv.x * xv.x + xv.y * xv.y + xv.z * xv.z + xv.w * xv.w;
    #pragma unroll
    for (int off = 16; off; off >>= 1) {
        s  += __shfl_down(s, off, 32);
        s2 += __shfl_down(s2, off, 32);
    }
    int w = t >> 5, ln = t & 31;
    if (ln == 0) { red0[w] = s; red1[w] = s2; }
    __syncthreads();
    float S = 0.f, S2 = 0.f;
    #pragma unroll
    for (int i = 0; i < 8; ++i) { S += red0[i]; S2 += red1[i]; }
    float mean = S * (1.f / DIMC);
    float var  = S2 * (1.f / DIMC) - mean * mean;
    float rs   = rsqrtf(var + 1e-5f);
    union { uint16_t h[4]; uint2 u; } o;
    o.h[0] = f2bf((xv.x - mean) * rs); o.h[1] = f2bf((xv.y - mean) * rs);
    o.h[2] = f2bf((xv.z - mean) * rs); o.h[3] = f2bf((xv.w - mean) * rs);
    *(uint2*)&xn[(size_t)row * DIMC + t * 4] = o.u;
}

// ================= shared GEMM mainloop =================
// 128x128x32 tiles, 8 waves (4x2), double-buffered LDS, register-staged
// global loads overlapping WMMA compute. One barrier per k-iteration.
// A logically (M x K) bf16 row-major, split at 'ksplit' between A0/A1.
// Bw is weight (N x K) bf16 row-major; computes C = A * Bw^T.
#define LSTR 40
#define LBUF (128 * LSTR)
__device__ __forceinline__ void gemm_tiles(
    const uint16_t* __restrict__ A0, int lda0, int ksplit,
    const uint16_t* __restrict__ A1, int lda1,
    const uint16_t* __restrict__ Bw, int ldb, int K,
    int mBase, int nBase,
    uint16_t* lA, uint16_t* lB, f32x8 acc[2][4]) {
    const int t    = threadIdx.x;
    const int lr   = t >> 1;          // row within 128-tile
    const int lh   = t & 1;           // which 16-elem half of k-tile
    const int lane = t & 31, wid = t >> 5;
    const int wm   = wid & 3, wn = wid >> 2;   // 4x2 wave grid
    const int m    = lane & 15, hh = lane >> 4;

    // prologue: stage tile 0 into registers
    uint4 av0, av1, bv0, bv1;
    {
        int kk = lh * 16;
        const uint16_t* sa = (kk < ksplit)
            ? (A0 + (size_t)(mBase + lr) * lda0 + kk)
            : (A1 + (size_t)(mBase + lr) * lda1 + (kk - ksplit));
        const uint16_t* sb = Bw + (size_t)(nBase + lr) * ldb + kk;
        av0 = *(const uint4*)sa; av1 = *(const uint4*)(sa + 8);
        bv0 = *(const uint4*)sb; bv1 = *(const uint4*)(sb + 8);
    }
    int buf = 0;
    for (int kt = 0; kt < K; kt += 32) {
        uint16_t* LA = lA + buf * LBUF;
        uint16_t* LB = lB + buf * LBUF;
        *(uint4*)&LA[lr * LSTR + lh * 16]     = av0;
        *(uint4*)&LA[lr * LSTR + lh * 16 + 8] = av1;
        *(uint4*)&LB[lr * LSTR + lh * 16]     = bv0;
        *(uint4*)&LB[lr * LSTR + lh * 16 + 8] = bv1;
        __syncthreads();
        if (kt + 32 < K) {            // stage next tile while this one computes
            int kk = kt + 32 + lh * 16;
            const uint16_t* sa = (kk < ksplit)
                ? (A0 + (size_t)(mBase + lr) * lda0 + kk)
                : (A1 + (size_t)(mBase + lr) * lda1 + (kk - ksplit));
            const uint16_t* sb = Bw + (size_t)(nBase + lr) * ldb + kk;
            av0 = *(const uint4*)sa; av1 = *(const uint4*)(sa + 8);
            bv0 = *(const uint4*)sb; bv1 = *(const uint4*)(sb + 8);
            __builtin_prefetch(sa + 32, 0, 1);   // global_prefetch_b8
            __builtin_prefetch(sb + 32, 0, 1);
        }
        // load ALL fragments into distinct registers, then fire 8 WMMAs
        bf16x16 af[2], bfr[4];
        #pragma unroll
        for (int i = 0; i < 2; ++i) {
            int r = wm * 32 + i * 16 + m;   // A-frag: [m][hh*8], [m][16+hh*8]
            af[i] = ld_frag(&LA[r * LSTR + hh * 8], &LA[r * LSTR + 16 + hh * 8]);
        }
        #pragma unroll
        for (int j = 0; j < 4; ++j) {
            int r = wn * 64 + j * 16 + m;   // B-frag: [n][hh*8], [n][16+hh*8]
            bfr[j] = ld_frag(&LB[r * LSTR + hh * 8], &LB[r * LSTR + 16 + hh * 8]);
        }
        #pragma unroll
        for (int j = 0; j < 4; ++j)
            #pragma unroll
            for (int i = 0; i < 2; ++i)
                acc[i][j] = wmma_bf16(af[i], bfr[j], acc[i][j]);
        buf ^= 1;
    }
}

// ================= GEMM1: h = xn * Win^T, epilogue scatters xm / q / k / v ======
__global__ __launch_bounds__(256, 1) void gemm1_kernel(
    const uint16_t* __restrict__ xn, const uint16_t* __restrict__ Win,
    uint16_t* __restrict__ hmlp, uint16_t* __restrict__ qb,
    uint16_t* __restrict__ kb, uint16_t* __restrict__ vb) {
    __shared__ uint16_t lA[2 * LBUF];
    __shared__ uint16_t lB[2 * LBUF];
    f32x8 acc[2][4];
    #pragma unroll
    for (int i = 0; i < 2; ++i)
        #pragma unroll
        for (int j = 0; j < 4; ++j) acc[i][j] = zerof8();
    int mBase = blockIdx.y * 128, nBase = blockIdx.x * 128;
    gemm_tiles(xn, DIMC, 1 << 30, nullptr, 0, Win, DIMC, DIMC, mBase, nBase, lA, lB, acc);
    int lane = threadIdx.x & 31, wid = threadIdx.x >> 5;
    int wm = wid & 3, wn = wid >> 2;
    int n = lane & 15, h2 = lane >> 4;
    #pragma unroll
    for (int i = 0; i < 2; ++i)
        #pragma unroll
        for (int j = 0; j < 4; ++j)
            #pragma unroll
            for (int e = 0; e < 8; ++e) {
                int gm = mBase + wm * 32 + i * 16 + h2 * 8 + e;
                int gn = nBase + wn * 64 + j * 16 + n;
                float val = acc[i][j][e];
                if (gn < MLPC) {
                    hmlp[(size_t)gm * MLPC + gn] = f2bf(val);
                } else {
                    int c = gn - MLPC;
                    int which = c >> 10;        // 0=q 1=k 2=v
                    int within = c & 1023;
                    int head = within >> 6, d = within & 63;
                    int bb = gm >> 10, nn = gm & 1023;
                    uint16_t* dst = (which == 0) ? qb : ((which == 1) ? kb : vb);
                    dst[(((size_t)(bb * HEADS + head)) * SEQ + nn) * HD + d] = f2bf(val);
                }
            }
}

// ================= flash attention: per (b,h,128-query tile) =================
__global__ __launch_bounds__(256, 1) void attn_kernel(
    const uint16_t* __restrict__ qg, const uint16_t* __restrict__ kg,
    const uint16_t* __restrict__ vg, uint16_t* __restrict__ xa) {
    __shared__ uint16_t Qs[128 * 72];
    __shared__ uint16_t Ks[32 * 72];
    __shared__ uint16_t Vt[64 * 40];          // V transposed: [d][key]
    __shared__ uint16_t Ps[8 * 16 * 32];      // per-wave P tile C->A relayout
    int qtile = blockIdx.x, head = blockIdx.y, b = blockIdx.z;
    int t = threadIdx.x;
    int lane = t & 31, w = t >> 5;
    size_t bh = (size_t)(b * HEADS + head);
    const uint16_t* Qp = qg + (bh * SEQ + qtile * 128) * HD;
    const uint16_t* Kp = kg + bh * SEQ * HD;
    const uint16_t* Vp = vg + bh * SEQ * HD;
    // stage Q tile (128x64)
    #pragma unroll
    for (int it = 0; it < 4; ++it) {
        int vec = t + it * 256;
        int r = vec >> 3, c = (vec & 7) * 8;
        *(uint4*)&Qs[r * 72 + c] = *(const uint4*)&Qp[(size_t)r * HD + c];
    }
    __syncthreads();
    int m = lane & 15, hh = lane >> 4;
    bf16x16 qf[2];
    #pragma unroll
    for (int i = 0; i < 2; ++i) {
        int r = w * 16 + m;
        qf[i] = ld_frag(&Qs[r * 72 + i * 32 + hh * 8], &Qs[r * 72 + i * 32 + 16 + hh * 8]);
    }
    f32x8 of[4];
    #pragma unroll
    for (int dj = 0; dj < 4; ++dj) of[dj] = zerof8();
    float mrun[8], lrun[8];
    #pragma unroll
    for (int e = 0; e < 8; ++e) { mrun[e] = -1e30f; lrun[e] = 0.f; }
    const float sc = 0.125f;   // HD^-0.5

    for (int kc = 0; kc < SEQ; kc += 32) {
        __syncthreads();
        {   // stage K chunk (32x64) row-major and V chunk transposed (64x32)
            int r = t >> 3, c = (t & 7) * 8;
            *(uint4*)&Ks[r * 72 + c] = *(const uint4*)&Kp[(size_t)(kc + r) * HD + c];
            uint4 vv = *(const uint4*)&Vp[(size_t)(kc + r) * HD + c];
            const uint16_t* ve = (const uint16_t*)&vv;
            #pragma unroll
            for (int e = 0; e < 8; ++e) Vt[(c + e) * 40 + r] = ve[e];
        }
        __syncthreads();
        // S = Q * K^T for 32 keys -> two 16x16 C fragments
        // load all 4 K-fragments first (distinct regs), then 4 WMMAs
        bf16x16 kf[4];
        #pragma unroll
        for (int j = 0; j < 2; ++j) {
            int r = j * 16 + m;   // key row (lane is B-fragment n)
            kf[j * 2 + 0] = ld_frag(&Ks[r * 72 + hh * 8], &Ks[r * 72 + 16 + hh * 8]);
            kf[j * 2 + 1] = ld_frag(&Ks[r * 72 + 32 + hh * 8], &Ks[r * 72 + 32 + 16 + hh * 8]);
        }
        f32x8 s0 = zerof8(), s1 = zerof8();
        s0 = wmma_bf16(qf[0], kf[0], s0);
        s0 = wmma_bf16(qf[1], kf[1], s0);
        s1 = wmma_bf16(qf[0], kf[2], s1);
        s1 = wmma_bf16(qf[1], kf[3], s1);
        // online softmax; rows of a C fragment live inside one 16-lane half
        float fac[8];
        #pragma unroll
        for (int e = 0; e < 8; ++e) {
            float a = fmaxf(s0[e], s1[e]) * sc;
            #pragma unroll
            for (int off = 1; off < 16; off <<= 1) a = fmaxf(a, __shfl_xor(a, off, 32));
            float mn = fmaxf(mrun[e], a);
            float f  = __expf(mrun[e] - mn);
            float p0 = __expf(s0[e] * sc - mn);
            float p1 = __expf(s1[e] * sc - mn);
            float ss = p0 + p1;
            #pragma unroll
            for (int off = 1; off < 16; off <<= 1) ss += __shfl_xor(ss, off, 32);
            lrun[e] = lrun[e] * f + ss;
            mrun[e] = mn;
            fac[e]  = f;
            int prow = e + 8 * hh;            // C layout: m = v + 8*(lane/16)
            Ps[w * 512 + prow * 32 + m]      = f2bf(p0);
            Ps[w * 512 + prow * 32 + 16 + m] = f2bf(p1);
        }
        #pragma unroll
        for (int dj = 0; dj < 4; ++dj)
            #pragma unroll
            for (int e = 0; e < 8; ++e) of[dj][e] = of[dj][e] * fac[e];
        __syncthreads();
        // reload P in A layout; load all 4 V-fragments, then 4 WMMAs
        bf16x16 pf = ld_frag(&Ps[w * 512 + m * 32 + hh * 8], &Ps[w * 512 + m * 32 + 16 + hh * 8]);
        bf16x16 vf[4];
        #pragma unroll
        for (int dj = 0; dj < 4; ++dj) {
            int r = dj * 16 + m;              // d row of Vt (lane is B-fragment n)
            vf[dj] = ld_frag(&Vt[r * 40 + hh * 8], &Vt[r * 40 + 16 + hh * 8]);
        }
        #pragma unroll
        for (int dj = 0; dj < 4; ++dj)
            of[dj] = wmma_bf16(pf, vf[dj], of[dj]);
    }
    // normalize and write xa in (token, head*64+d) layout
    #pragma unroll
    for (int dj = 0; dj < 4; ++dj)
        #pragma unroll
        for (int e = 0; e < 8; ++e) {
            int row = e + 8 * hh;
            size_t tok = (size_t)b * SEQ + qtile * 128 + w * 16 + row;
            int col = head * HD + dj * 16 + m;
            xa[tok * DIMC + col] = f2bf(of[dj][e] / lrun[e]);
        }
}

// ================= GEMM2: o = [hmlp | xa] * Wout^T -> fp32 =================
__global__ __launch_bounds__(256, 1) void gemm2_kernel(
    const uint16_t* __restrict__ hmlp, const uint16_t* __restrict__ xa,
    const uint16_t* __restrict__ Wout, float* __restrict__ ob) {
    __shared__ uint16_t lA[2 * LBUF];
    __shared__ uint16_t lB[2 * LBUF];
    f32x8 acc[2][4];
    #pragma unroll
    for (int i = 0; i < 2; ++i)
        #pragma unroll
        for (int j = 0; j < 4; ++j) acc[i][j] = zerof8();
    int mBase = blockIdx.y * 128, nBase = blockIdx.x * 128;
    gemm_tiles(hmlp, MLPC, MLPC, xa, DIMC, Wout, OUT_IN, OUT_IN, mBase, nBase, lA, lB, acc);
    int lane = threadIdx.x & 31, wid = threadIdx.x >> 5;
    int wm = wid & 3, wn = wid >> 2;
    int n = lane & 15, h2 = lane >> 4;
    #pragma unroll
    for (int i = 0; i < 2; ++i)
        #pragma unroll
        for (int j = 0; j < 4; ++j)
            #pragma unroll
            for (int e = 0; e < 8; ++e) {
                int gm = mBase + wm * 32 + i * 16 + h2 * 8 + e;
                int gn = nBase + wn * 64 + j * 16 + n;
                ob[(size_t)gm * OUT_OUT + gn] = acc[i][j][e];
            }
}

// ================= final: out = o_mlp + o_attn + residual =================
__global__ __launch_bounds__(256) void final_kernel(const float* __restrict__ ob,
                                                    const float* __restrict__ x,
                                                    float* __restrict__ out) {
    size_t v4 = (size_t)blockIdx.x * 256 + threadIdx.x;
    size_t base = v4 * 4;
    size_t row = base >> 10;
    int col = (int)(base & 1023);
    float4 o1 = *(const float4*)&ob[row * OUT_OUT + col];
    float4 o2 = *(const float4*)&ob[row * OUT_OUT + 1024 + col];
    float4 xr = *(const float4*)&x[base];
    float4 r;
    r.x = o1.x + o2.x + xr.x; r.y = o1.y + o2.y + xr.y;
    r.z = o1.z + o2.z + xr.z; r.w = o1.w + o2.w + xr.w;
    *(float4*)&out[base] = r;
}

// ================= launch =================
extern "C" void kernel_launch(void* const* d_in, const int* in_sizes, int n_in,
                              void* d_out, int out_size, void* d_ws, size_t ws_size,
                              hipStream_t stream) {
    const float* x       = (const float*)d_in[0];
    const float* in_cb   = (const float*)d_in[1];
    const int*   in_idx  = (const int*)d_in[2];
    const float* out_cb  = (const float*)d_in[3];
    const int*   out_idx = (const int*)d_in[4];
    float* out = (float*)d_out;

    uintptr_t p = (uintptr_t)d_ws;
    auto carve = [&](size_t bytes) -> void* {
        void* r = (void*)p;
        p += (bytes + 255) & ~(size_t)255;
        return r;
    };
    uint16_t* Win  = (uint16_t*)carve((size_t)IN_OUT * DIMC * 2);
    uint16_t* Wout = (uint16_t*)carve((size_t)OUT_OUT * OUT_IN * 2);
    uint16_t* xn   = (uint16_t*)carve((size_t)NTOK * DIMC * 2);
    uint16_t* hmlp = (uint16_t*)carve((size_t)NTOK * MLPC * 2);
    uint16_t* qb   = (uint16_t*)carve((size_t)NTOK * DIMC * 2);
    uint16_t* kb   = (uint16_t*)carve((size_t)NTOK * DIMC * 2);
    uint16_t* vb   = (uint16_t*)carve((size_t)NTOK * DIMC * 2);
    uint16_t* xab  = (uint16_t*)carve((size_t)NTOK * DIMC * 2);
    float*    ob   = (float*)carve((size_t)NTOK * OUT_OUT * 4);

    decode_win_kernel <<<IN_OUT,  256, 0, stream>>>(in_cb, in_idx, Win);
    decode_wout_kernel<<<OUT_OUT, 256, 0, stream>>>(out_cb, out_idx, Wout);
    ln_kernel         <<<NTOK,    256, 0, stream>>>(x, xn);
    gemm1_kernel      <<<dim3(IN_OUT / 128, NTOK / 128), 256, 0, stream>>>(xn, Win, hmlp, qb, kb, vb);
    attn_kernel       <<<dim3(SEQ / 128, HEADS, 4), 256, 0, stream>>>(qb, kb, vb, xab);
    gemm2_kernel      <<<dim3(OUT_OUT / 128, NTOK / 128), 256, 0, stream>>>(hmlp, xab, Wout, ob);
    final_kernel      <<<(NTOK * DIMC) / (256 * 4), 256, 0, stream>>>(ob, x, out);
}
